// Linear4bit_65738769432994
// MI455X (gfx1250) — matrix operand order repaired and checked by reading, not compiled, not run on hardware
//
#include <hip/hip_runtime.h>

typedef __attribute__((ext_vector_type(16))) _Float16 v16h;
typedef __attribute__((ext_vector_type(8)))  _Float16 v8h;
typedef __attribute__((ext_vector_type(4)))  _Float16 v4h;
typedef __attribute__((ext_vector_type(8)))  float    v8f;
typedef __attribute__((ext_vector_type(4)))  float    v4f;
typedef __attribute__((ext_vector_type(4)))  int      v4i;

// bitsandbytes NF4 codebook
__device__ __constant__ float NF4_TAB[16] = {
    -1.0f, -0.6961928009986877f, -0.5250730514526367f, -0.39491748809814453f,
    -0.28444138169288635f, -0.18477343022823334f, -0.09105003625154495f, 0.0f,
    0.07958029955625534f, 0.16093020141124725f, 0.24611230194568634f,
    0.33791524171829224f, 0.44070982933044434f, 0.5626170039176941f,
    0.7229568362236023f, 1.0f};

__device__ __forceinline__ v16h cat8(v8h lo, v8h hi) {
    return __builtin_shufflevector(lo, hi,
        0,1,2,3,4,5,6,7,8,9,10,11,12,13,14,15);
}

// ---------------------------------------------------------------------------
// Pass 0: x f32 -> f16 (8 elements / thread)
// ---------------------------------------------------------------------------
__global__ __launch_bounds__(256) void cvt_f32_f16(
    const float* __restrict__ in, _Float16* __restrict__ out, long n)
{
    long base = ((long)blockIdx.x * 256 + threadIdx.x) * 8L;
    if (base >= n) return;
    v4f a = *(const v4f*)(in + base);
    v4f b = *(const v4f*)(in + base + 4);
    v8h o;
    o[0] = (_Float16)a.x; o[1] = (_Float16)a.y;
    o[2] = (_Float16)a.z; o[3] = (_Float16)a.w;
    o[4] = (_Float16)b.x; o[5] = (_Float16)b.y;
    o[6] = (_Float16)b.z; o[7] = (_Float16)b.w;
    *(v8h*)(out + base) = o;
}

// ---------------------------------------------------------------------------
// Pass 1: NF4 (int32 codes) + per-64 absmax -> f16 weights in workspace.
// ---------------------------------------------------------------------------
__global__ __launch_bounds__(256) void nf4_dequant_f16(
    const int* __restrict__ codes, const float* __restrict__ absmax,
    _Float16* __restrict__ w, long total)
{
    __shared__ float lut[16];
    if (threadIdx.x < 16) lut[threadIdx.x] = NF4_TAB[threadIdx.x];
    __syncthreads();

    long base = ((long)blockIdx.x * 256 + threadIdx.x) * 16L;
    if (base >= total) return;

    float amax = absmax[base >> 6];   // BLOCK = 64
    v4i c0 = *(const v4i*)(codes + base);
    v4i c1 = *(const v4i*)(codes + base + 4);
    v4i c2 = *(const v4i*)(codes + base + 8);
    v4i c3 = *(const v4i*)(codes + base + 12);

    v8h o0, o1;
    o0[0] = (_Float16)(lut[c0.x & 15] * amax);
    o0[1] = (_Float16)(lut[c0.y & 15] * amax);
    o0[2] = (_Float16)(lut[c0.z & 15] * amax);
    o0[3] = (_Float16)(lut[c0.w & 15] * amax);
    o0[4] = (_Float16)(lut[c1.x & 15] * amax);
    o0[5] = (_Float16)(lut[c1.y & 15] * amax);
    o0[6] = (_Float16)(lut[c1.z & 15] * amax);
    o0[7] = (_Float16)(lut[c1.w & 15] * amax);
    o1[0] = (_Float16)(lut[c2.x & 15] * amax);
    o1[1] = (_Float16)(lut[c2.y & 15] * amax);
    o1[2] = (_Float16)(lut[c2.z & 15] * amax);
    o1[3] = (_Float16)(lut[c2.w & 15] * amax);
    o1[4] = (_Float16)(lut[c3.x & 15] * amax);
    o1[5] = (_Float16)(lut[c3.y & 15] * amax);
    o1[6] = (_Float16)(lut[c3.z & 15] * amax);
    o1[7] = (_Float16)(lut[c3.w & 15] * amax);

    *(v8h*)(w + base)     = o0;
    *(v8h*)(w + base + 8) = o1;
}

// ---------------------------------------------------------------------------
// Pass 2: WMMA GEMM  out[M][N] = X[M][K] * W[N][K]^T + bias[N]
// 256 threads = 8 wave32 (2 M x 4 N), block tile 128x256, K tile 32,
// double-buffered LDS filled with global_load_async_to_lds_b128 (ASYNCcnt).
// Each wave: 4x4 accumulators of v_wmma_f32_16x16x32_f16 (64x64 region).
// AT = _Float16 (pre-converted x, fully async) or float (reg staging + cvt).
// ---------------------------------------------------------------------------
#define TM 128
#define TN 256
#define TK 32
#define LDP 40   // padded LDS row stride in halfs (80 B -> 20-bank skew)

template <typename AT>
__global__ __launch_bounds__(256) void nf4_wmma_gemm(
    const AT* __restrict__ X, const _Float16* __restrict__ W,
    const float* __restrict__ bias, float* __restrict__ out,
    int M, int N, int K)
{
    constexpr bool A16 = (sizeof(AT) == 2);

    __shared__ _Float16 As[2][TM * LDP];   // 2 x 10 KB
    __shared__ _Float16 Bs[2][TN * LDP];   // 2 x 20 KB

    const int tid  = threadIdx.x;
    const int lane = tid & 31;
    const int wave = tid >> 5;
    const int waveM = (wave & 1) * 64;     // 2 waves along M
    const int waveN = (wave >> 1) * 64;    // 4 waves along N
    const int blockN0 = blockIdx.x * TN;
    const int blockM0 = blockIdx.y * TM;

    // fragment lane decomposition (ISA 7.12.2)
    const int h = lane >> 4;
    const int r = lane & 15;

    // async staging map: 16 B / thread, 4 threads per 32-half row, 64 rows/pass
    const int sRow = tid >> 2;             // 0..63
    const int sCol = (tid & 3) * 8;        // halfs
    // f32-A staging map: float4 / thread, 8 threads/row, 32 rows/pass
    const int aRow32 = tid >> 3;
    const int aCol32 = (tid & 7) * 4;

    v8f zero = {};
    v8f acc[4][4];
#pragma unroll
    for (int mt = 0; mt < 4; ++mt)
#pragma unroll
        for (int nt = 0; nt < 4; ++nt) acc[mt][nt] = zero;

    v4f aReg[4];   // used only when AT == float

    auto issue_fills = [&](int kt, int buf) {
        // B tile: 256 rows x 32 halfs, 4 async b128 per thread
#pragma unroll
        for (int p = 0; p < 4; ++p) {
            const _Float16* g = W + (long)(blockN0 + p * 64 + sRow) * K + kt + sCol;
            unsigned l = (unsigned)(unsigned long long)&Bs[buf][(p * 64 + sRow) * LDP + sCol];
            asm volatile("global_load_async_to_lds_b128 %0, %1, off"
                         :: "v"(l), "v"(g) : "memory");
        }
        if constexpr (A16) {
            // A tile: 128 rows x 32 halfs, 2 async b128 per thread
#pragma unroll
            for (int p = 0; p < 2; ++p) {
                const _Float16* g = X + (long)(blockM0 + p * 64 + sRow) * K + kt + sCol;
                unsigned l = (unsigned)(unsigned long long)&As[buf][(p * 64 + sRow) * LDP + sCol];
                asm volatile("global_load_async_to_lds_b128 %0, %1, off"
                             :: "v"(l), "v"(g) : "memory");
            }
        } else {
#pragma unroll
            for (int p = 0; p < 4; ++p)
                aReg[p] = *(const v4f*)((const AT*)X + (long)(blockM0 + p * 32 + aRow32) * K + kt + aCol32);
        }
    };

    auto store_a_regs = [&](int buf) {     // f32-A path only
#pragma unroll
        for (int p = 0; p < 4; ++p) {
            v4h t;
            t[0] = (_Float16)aReg[p].x;
            t[1] = (_Float16)aReg[p].y;
            t[2] = (_Float16)aReg[p].z;
            t[3] = (_Float16)aReg[p].w;
            *(v4h*)&As[buf][(p * 32 + aRow32) * LDP + aCol32] = t;
        }
    };

    // prologue: fill buffer 0 with K-tile 0
    issue_fills(0, 0);
    if constexpr (!A16) store_a_regs(0);

    int p = 0;
    for (int k0 = 0; k0 < K; k0 += TK) {
        asm volatile("s_wait_asynccnt 0x0" ::: "memory");
        __syncthreads();   // fills of buf p (and ds stores) visible to all waves

        const int kn = k0 + TK;
        if (kn < K) issue_fills(kn, p ^ 1);   // overlap with WMMAs below

        // A fragments: lane half h holds K {8h..8h+7, 16+8h..16+8h+7}
        v16h aF[4];
#pragma unroll
        for (int mt = 0; mt < 4; ++mt) {
            const _Float16* s = &As[p][(waveM + mt * 16 + r) * LDP];
            v8h lo = *(const v8h*)(s + 8 * h);
            v8h hi = *(const v8h*)(s + 16 + 8 * h);
            aF[mt] = cat8(lo, hi);
        }
        // B fragments: lane half h holds K {8h..8h+7, 16+8h..16+8h+7}
        v16h bF[4];
#pragma unroll
        for (int nt = 0; nt < 4; ++nt) {
            const _Float16* s = &Bs[p][(waveN + nt * 16 + r) * LDP];
            v8h lo = *(const v8h*)(s + 8 * h);
            v8h hi = *(const v8h*)(s + 16 + 8 * h);
            bF[nt] = cat8(lo, hi);
        }

#pragma unroll
        for (int mt = 0; mt < 4; ++mt)
#pragma unroll
            for (int nt = 0; nt < 4; ++nt)
                acc[mt][nt] = __builtin_amdgcn_wmma_f32_16x16x32_f16(
                    false, aF[mt], false, bF[nt],
                    (short)0, acc[mt][nt], false, false);

        if constexpr (!A16) { if (kn < K) store_a_regs(p ^ 1); }
        p ^= 1;
    }

    // epilogue: C/D layout — lane%16 = N, M = 8*(lane/16) + vgpr
#pragma unroll
    for (int nt = 0; nt < 4; ++nt) {
        const int col = blockN0 + waveN + nt * 16 + r;
        const float bv = bias[col];
#pragma unroll
        for (int mt = 0; mt < 4; ++mt) {
            const int row0 = blockM0 + waveM + mt * 16 + 8 * h;
#pragma unroll
            for (int v = 0; v < 8; ++v)
                out[(long)(row0 + v) * N + col] = acc[mt][nt][v] + bv;
        }
    }
}

// ---------------------------------------------------------------------------
extern "C" void kernel_launch(void* const* d_in, const int* in_sizes, int n_in,
                              void* d_out, int out_size, void* d_ws, size_t ws_size,
                              hipStream_t stream) {
    const float* x      = (const float*)d_in[0];
    const int*   codes  = (const int*)d_in[1];
    const float* absmax = (const float*)d_in[2];
    const float* bias   = (const float*)d_in[3];
    float*       out    = (float*)d_out;

    const int  out_f = in_sizes[3];                  // 4096
    const long wtot  = (long)in_sizes[1];            // out_f * in_f
    const int  in_f  = (int)(wtot / out_f);          // 4096
    const long xtot  = (long)in_sizes[0];            // M * in_f
    const int  M     = (int)(xtot / in_f);           // 8192

    const size_t wbytes = (size_t)wtot * 2;
    const size_t xbytes = (size_t)xtot * 2;

    _Float16* w16 = (_Float16*)d_ws;

    // Pass 1: dequantize NF4 -> f16 (16 elems / thread)
    dim3 dqGrid((unsigned)((wtot / 16 + 255) / 256));
    nf4_dequant_f16<<<dqGrid, 256, 0, stream>>>(codes, absmax, w16, wtot);

    dim3 grid(out_f / TN, M / TM);

    if (ws_size >= wbytes + xbytes) {
        // Pass 0: convert x to f16 so both tiles use async-to-LDS
        _Float16* x16 = (_Float16*)((char*)d_ws + wbytes);
        dim3 cvGrid((unsigned)((xtot / 8 + 255) / 256));
        cvt_f32_f16<<<cvGrid, 256, 0, stream>>>(x, x16, xtot);
        nf4_wmma_gemm<_Float16><<<grid, 256, 0, stream>>>(
            x16, w16, bias, out, M, out_f, in_f);
    } else {
        nf4_wmma_gemm<float><<<grid, 256, 0, stream>>>(
            x, w16, bias, out, M, out_f, in_f);
    }
}
